// CausalSelfAttention_28183575396613
// MI455X (gfx1250) — matrix operand order repaired and checked by reading, not compiled, not run on hardware
//
#include <hip/hip_runtime.h>

#define B_ 4
#define T_ 2048
#define C_ 1024
#define H_ 16
#define D_ 64

typedef __bf16 bf16;
typedef __attribute__((ext_vector_type(16))) __bf16 v16bf;
typedef __attribute__((ext_vector_type(8)))  float  v8f;

union Frag { unsigned u[8]; v16bf v; };

#define INFF __builtin_inff()

__device__ __forceinline__ bf16 f2bf(float f) {
  unsigned u = __builtin_bit_cast(unsigned, f);
  unsigned r = u + 0x7FFFu + ((u >> 16) & 1u);   // round-to-nearest-even
  unsigned short h = (unsigned short)(r >> 16);
  return __builtin_bit_cast(bf16, h);
}

__device__ __forceinline__ v8f vzero() {
  v8f z = {0.f,0.f,0.f,0.f,0.f,0.f,0.f,0.f};
  return z;
}

// one 16B chunk: global -> LDS via CDNA5 async DMA path (ASYNCcnt)
__device__ __forceinline__ void async_b128(const void* gaddr, void* lds) {
  unsigned ldsoff = (unsigned)(size_t)lds;
  unsigned long long g = (unsigned long long)gaddr;
  asm volatile("global_load_async_to_lds_b128 %0, %1, off"
               :: "v"(ldsoff), "v"(g) : "memory");
}
// 64B: one address pair, IOFFSET advances both global and LDS sides
__device__ __forceinline__ void async_b128x4(const void* gaddr, void* lds) {
  unsigned ldsoff = (unsigned)(size_t)lds;
  unsigned long long g = (unsigned long long)gaddr;
  asm volatile("global_load_async_to_lds_b128 %0, %1, off\n\t"
               "global_load_async_to_lds_b128 %0, %1, off offset:16\n\t"
               "global_load_async_to_lds_b128 %0, %1, off offset:32\n\t"
               "global_load_async_to_lds_b128 %0, %1, off offset:48"
               :: "v"(ldsoff), "v"(g) : "memory");
}
__device__ __forceinline__ void async_b128x2(const void* gaddr, void* lds) {
  unsigned ldsoff = (unsigned)(size_t)lds;
  unsigned long long g = (unsigned long long)gaddr;
  asm volatile("global_load_async_to_lds_b128 %0, %1, off\n\t"
               "global_load_async_to_lds_b128 %0, %1, off offset:16"
               :: "v"(ldsoff), "v"(g) : "memory");
}
__device__ __forceinline__ void async_b64(const void* gaddr, void* lds) {
  unsigned ldsoff = (unsigned)(size_t)lds;
  unsigned long long g = (unsigned long long)gaddr;
  asm volatile("global_load_async_to_lds_b64 %0, %1, off"
               :: "v"(ldsoff), "v"(g) : "memory");
}
__device__ __forceinline__ void wait_async0() {
  asm volatile("s_wait_asynccnt 0x0" ::: "memory");
}

// ---------------------------------------------------------------- converts
__global__ __launch_bounds__(256) void f32_to_bf16_kernel(
    const float* __restrict__ in, bf16* __restrict__ out, int n) {
  int i = blockIdx.x * 256 + threadIdx.x;
  if (i < n) out[i] = f2bf(in[i]);
}

// W[k][n] fp32 -> Wt[n][k] bf16 (transpose once so GEMM staging is vectorized)
__global__ __launch_bounds__(256) void f32_to_bf16_T_kernel(
    const float* __restrict__ in, bf16* __restrict__ out) {
  __shared__ float tile[32][33];
  const int k0 = blockIdx.x * 32, n0 = blockIdx.y * 32;
  const int tx = threadIdx.x & 31, ty = threadIdx.x >> 5;   // 32x8
  for (int r = ty; r < 32; r += 8)
    tile[r][tx] = in[(size_t)(k0 + r) * C_ + n0 + tx];
  __syncthreads();
  for (int r = ty; r < 32; r += 8)
    out[(size_t)(n0 + r) * C_ + k0 + tx] = f2bf(tile[tx][r]);
}

// ---------------------------------------------------------------- GEMMs
// X[8192,1024](bf16) x W (given as Wt[n][k] bf16) + b
#define BM 64
#define BN 128
#define BK 32
#define XPAD 40   // bank-conflict-free row stride, keeps 16B alignment
#define WPAD 40

__global__ __launch_bounds__(256) void qkv_gemm_kernel(
    const bf16* __restrict__ X,
    const bf16* __restrict__ Wq, const bf16* __restrict__ Wk, const bf16* __restrict__ Wv,
    const float* __restrict__ bq, const float* __restrict__ bk, const float* __restrict__ bv,
    bf16* __restrict__ Qo, bf16* __restrict__ Ko, bf16* __restrict__ Vo)
{
  __shared__ bf16 Xs[2][BM][XPAD];
  __shared__ bf16 Ws[2][BN][WPAD];
  const int tid = threadIdx.x, w = tid >> 5, lane = tid & 31;
  const int m0 = blockIdx.y * BM, n0 = blockIdx.x * BN;
  const bf16* Wt; const float* bias; bf16* outp;
  if (blockIdx.z == 0)      { Wt = Wq; bias = bq; outp = Qo; }
  else if (blockIdx.z == 1) { Wt = Wk; bias = bk; outp = Ko; }
  else                      { Wt = Wv; bias = bv; outp = Vo; }

  const int wm = (w >> 2) * 32, wn = (w & 3) * 32;
  const int halfA = lane >> 4, mn = lane & 15;
  v8f acc[2][2] = { { vzero(), vzero() }, { vzero(), vzero() } };

  const int xr = tid >> 2, xc = (tid & 3) * 8;
  const bf16* xsrc = X + (size_t)(m0 + xr) * C_ + xc;
  const int wr0 = tid >> 2, wk0 = (tid & 3) * 8;          // pass 0
  const int wr1 = (256 + tid) >> 2, wk1 = wk0;            // pass 1

  auto stage = [&](int k0, int bs) {
    async_b128(xsrc + k0, &Xs[bs][xr][xc]);
    async_b128(Wt + (size_t)(n0 + wr0) * C_ + k0 + wk0, &Ws[bs][wr0][wk0]);
    async_b128(Wt + (size_t)(n0 + wr1) * C_ + k0 + wk1, &Ws[bs][wr1][wk1]);
  };

  stage(0, 0);
  wait_async0();
  __syncthreads();

  const int NT = C_ / BK;
  for (int it = 0; it < NT; ++it) {
    const int bs = it & 1;
    Frag A[2], Bf[2];
#pragma unroll
    for (int tm = 0; tm < 2; ++tm)
#pragma unroll
      for (int i = 0; i < 8; ++i) {
        int ka = (i < 4 ? 2 * i : 2 * i + 8) + halfA * 8;    // A 16x32 bf16 layout
        A[tm].u[i] = *(const unsigned*)&Xs[bs][wm + tm * 16 + mn][ka];
      }
#pragma unroll
    for (int tn = 0; tn < 2; ++tn)
#pragma unroll
      for (int i = 0; i < 8; ++i) {
        int kb = (i < 4 ? 2 * i : 2 * i + 8) + halfA * 8;    // B 32x16 bf16 layout (same k order as A)
        Bf[tn].u[i] = *(const unsigned*)&Ws[bs][wn + tn * 16 + mn][kb];
      }
    if (it + 1 < NT) {
      stage((it + 1) * BK, bs ^ 1);                          // async, overlaps WMMAs
      __builtin_prefetch(xsrc + (it + 2) * BK, 0, 1);        // global_prefetch_b8
    }
#pragma unroll
    for (int tm = 0; tm < 2; ++tm)
#pragma unroll
      for (int tn = 0; tn < 2; ++tn)
        acc[tm][tn] = __builtin_amdgcn_wmma_f32_16x16x32_bf16(
            false, A[tm].v, false, Bf[tn].v, (short)0, acc[tm][tn], false, false);
    wait_async0();                                           // publish before barrier
    __syncthreads();
  }

  // epilogue: +bias; Q,K -> [B,H,T,D]; V -> [B,H,D,T] (so P*V B-operand is contiguous)
  const bool vlay = (blockIdx.z == 2);
#pragma unroll
  for (int tn = 0; tn < 2; ++tn) {
    int c = n0 + wn + tn * 16 + mn;
    float bvv = bias[c];
    int hh = c >> 6, dd = c & 63;
#pragma unroll
    for (int tm = 0; tm < 2; ++tm)
#pragma unroll
      for (int r = 0; r < 8; ++r) {
        int gm = m0 + wm + tm * 16 + r + 8 * halfA;
        int bb = gm >> 11, tt = gm & (T_ - 1);
        bf16 val = f2bf(acc[tm][tn][r] + bvv);
        if (vlay)
          outp[(((size_t)bb * H_ + hh) * D_ + dd) * T_ + tt] = val;
        else
          outp[(((size_t)bb * H_ + hh) * T_ + tt) * D_ + dd] = val;
      }
  }
}

__global__ __launch_bounds__(256) void proj_gemm_kernel(
    const bf16* __restrict__ X, const bf16* __restrict__ Wt,
    const float* __restrict__ bias, float* __restrict__ outp)
{
  __shared__ bf16 Xs[2][BM][XPAD];
  __shared__ bf16 Ws[2][BN][WPAD];
  const int tid = threadIdx.x, w = tid >> 5, lane = tid & 31;
  const int m0 = blockIdx.y * BM, n0 = blockIdx.x * BN;
  const int wm = (w >> 2) * 32, wn = (w & 3) * 32;
  const int halfA = lane >> 4, mn = lane & 15;
  v8f acc[2][2] = { { vzero(), vzero() }, { vzero(), vzero() } };

  const int xr = tid >> 2, xc = (tid & 3) * 8;
  const bf16* xsrc = X + (size_t)(m0 + xr) * C_ + xc;
  const int wr0 = tid >> 2, wk0 = (tid & 3) * 8;
  const int wr1 = (256 + tid) >> 2, wk1 = wk0;

  auto stage = [&](int k0, int bs) {
    async_b128(xsrc + k0, &Xs[bs][xr][xc]);
    async_b128(Wt + (size_t)(n0 + wr0) * C_ + k0 + wk0, &Ws[bs][wr0][wk0]);
    async_b128(Wt + (size_t)(n0 + wr1) * C_ + k0 + wk1, &Ws[bs][wr1][wk1]);
  };

  stage(0, 0);
  wait_async0();
  __syncthreads();

  const int NT = C_ / BK;
  for (int it = 0; it < NT; ++it) {
    const int bs = it & 1;
    Frag A[2], Bf[2];
#pragma unroll
    for (int tm = 0; tm < 2; ++tm)
#pragma unroll
      for (int i = 0; i < 8; ++i) {
        int ka = (i < 4 ? 2 * i : 2 * i + 8) + halfA * 8;
        A[tm].u[i] = *(const unsigned*)&Xs[bs][wm + tm * 16 + mn][ka];
      }
#pragma unroll
    for (int tn = 0; tn < 2; ++tn)
#pragma unroll
      for (int i = 0; i < 8; ++i) {
        int kb = (i < 4 ? 2 * i : 2 * i + 8) + halfA * 8;
        Bf[tn].u[i] = *(const unsigned*)&Ws[bs][wn + tn * 16 + mn][kb];
      }
    if (it + 1 < NT) {
      stage((it + 1) * BK, bs ^ 1);
      __builtin_prefetch(xsrc + (it + 2) * BK, 0, 1);
    }
#pragma unroll
    for (int tm = 0; tm < 2; ++tm)
#pragma unroll
      for (int tn = 0; tn < 2; ++tn)
        acc[tm][tn] = __builtin_amdgcn_wmma_f32_16x16x32_bf16(
            false, A[tm].v, false, Bf[tn].v, (short)0, acc[tm][tn], false, false);
    wait_async0();
    __syncthreads();
  }

#pragma unroll
  for (int tn = 0; tn < 2; ++tn) {
    int c = n0 + wn + tn * 16 + mn;
    float bvv = bias[c];
#pragma unroll
    for (int tm = 0; tm < 2; ++tm)
#pragma unroll
      for (int r = 0; r < 8; ++r) {
        int gm = m0 + wm + tm * 16 + r + 8 * halfA;
        outp[(size_t)gm * C_ + c] = acc[tm][tn][r] + bvv;
      }
  }
}

// ---------------------------------------------------------------- fused attention
// One workgroup = (b, h, 16-row q tile). Scores live in LDS (CDNA5: 320KB/WGP).
#define SC_BYTES   (16 * T_ * 4)             // 131072: fp32 scores [16][2048]
#define PBS        2056                       // padded prob row stride (bf16)
#define PB_BYTES   (16 * PBS * 2)             // 65792
#define QS_STRIDE  72
#define QS_BYTES   (16 * QS_STRIDE * 2)       // 2304
#define KS_STRIDE  72
#define KS_BYTES   (8 * 16 * KS_STRIDE * 2)   // 18432: per-wave K tile
#define VSTRIDE    40
#define VS_BYTES   (8 * 16 * VSTRIDE * 2)     // 10240: per-wave Vt tile [d][k]
#define RED_BYTES  (256 * 4)
#define PART_BYTES (4 * 256 * 4)
#define ATTN_SMEM  (SC_BYTES + PB_BYTES + QS_BYTES + KS_BYTES + VS_BYTES + RED_BYTES + PART_BYTES)

__global__ __launch_bounds__(256) void attn_kernel(
    const bf16* __restrict__ Q, const bf16* __restrict__ K, const bf16* __restrict__ Vt,
    float* __restrict__ att, bf16* __restrict__ Yint)
{
  extern __shared__ char smem[];
  float* sc  = (float*)smem;
  bf16*  pb  = (bf16*)(smem + SC_BYTES);
  bf16*  qs  = (bf16*)(smem + SC_BYTES + PB_BYTES);
  bf16*  ks  = (bf16*)(smem + SC_BYTES + PB_BYTES + QS_BYTES);
  bf16*  vs  = (bf16*)(smem + SC_BYTES + PB_BYTES + QS_BYTES + KS_BYTES);
  float* red = (float*)(smem + SC_BYTES + PB_BYTES + QS_BYTES + KS_BYTES + VS_BYTES);
  float* part= (float*)(smem + SC_BYTES + PB_BYTES + QS_BYTES + KS_BYTES + VS_BYTES + RED_BYTES);

  const int tid = threadIdx.x, w = tid >> 5, lane = tid & 31;
  const int halfA = lane >> 4, mn = lane & 15;
  const int qt = blockIdx.x, h = blockIdx.y, b = blockIdx.z;
  const int q0 = qt * 16;
  const size_t bhT = ((size_t)b * H_ + h) * T_;
  const bf16* Qp = Q  + bhT * D_;
  const bf16* Kp = K  + bhT * D_;
  const bf16* Vp = Vt + bhT * D_;   // [B,H,D,T]: same byte offset for (b,h)
  float* attp = att + bhT * T_ + (size_t)q0 * T_;

  { // stage Q tile [16][64] via async DMA (8B per thread)
    int r = tid >> 4, c4 = (tid & 15) * 4;
    async_b64(&Qp[(size_t)(q0 + r) * D_ + c4], &qs[r * QS_STRIDE + c4]);
  }
  wait_async0();
  __syncthreads();

  const int nk = qt + 1;          // causal: only k-tiles <= q-tile
  const int nk16 = nk * 16;       // = q0 + 16

  // -------- phase 1: S = Q K^T (waves split k-tiles), mask+scale into LDS
  bf16* kw = ks + w * (16 * KS_STRIDE);
  for (int kt = w; kt < nk; kt += 8) {
    { // stage K tile [16][64] for this wave (64B per lane, contiguous both sides)
      int r = lane >> 1, c32 = (lane & 1) * 32;
      async_b128x4(&Kp[(size_t)(kt * 16 + r) * D_ + c32], &kw[r * KS_STRIDE + c32]);
    }
    if (kt + 8 < nk)
      __builtin_prefetch(&Kp[(size_t)((kt + 8) * 16 + (lane >> 1)) * D_ + (lane & 1) * 32], 0, 1);
    wait_async0();                 // this wave's tile is in LDS
    v8f s = vzero();
#pragma unroll
    for (int kk = 0; kk < 2; ++kk) {
      Frag A, Bf;
#pragma unroll
      for (int i = 0; i < 8; ++i) {
        int ka = (i < 4 ? 2 * i : 2 * i + 8) + halfA * 8 + kk * 32;
        A.u[i] = *(const unsigned*)&qs[mn * QS_STRIDE + ka];
        int kb = (i < 4 ? 2 * i : 2 * i + 8) + halfA * 8 + kk * 32;   // B[k][n] = Ktile[n][k]
        Bf.u[i] = *(const unsigned*)&kw[mn * KS_STRIDE + kb];
      }
      s = __builtin_amdgcn_wmma_f32_16x16x32_bf16(false, A.v, false, Bf.v, (short)0, s, false, false);
    }
#pragma unroll
    for (int r = 0; r < 8; ++r) {
      int m = r + 8 * halfA;
      int kg = kt * 16 + mn;
      sc[m * T_ + kg] = (kg <= q0 + m) ? s[r] * 0.125f : -INFF;  // 1/sqrt(64)
    }
  }
  __syncthreads();

  // -------- phase 2: row softmax (16 threads per row), write att once
  const int row = tid >> 4, sub = tid & 15;
  float* srow = sc + row * T_;
  float mx = -INFF;
  for (int k = sub; k < nk16; k += 16) mx = fmaxf(mx, srow[k]);
  red[tid] = mx;
  __syncthreads();
  float rmax = -INFF;
#pragma unroll
  for (int i = 0; i < 16; ++i) rmax = fmaxf(rmax, red[row * 16 + i]);
  __syncthreads();
  float sum = 0.f;
  for (int k = sub; k < nk16; k += 16) {
    float e = __expf(srow[k] - rmax);
    srow[k] = e;
    sum += e;
  }
  red[tid] = sum;
  __syncthreads();
  float rsum = 0.f;
#pragma unroll
  for (int i = 0; i < 16; ++i) rsum += red[row * 16 + i];
  const float inv = 1.0f / rsum;
  const int kvpad = (nk16 + 31) & ~31;   // pad probs to WMMA K=32
  float* arow = attp + (size_t)row * T_;
  bf16*  prow = pb + row * PBS;
  for (int k = sub; k < T_; k += 16) {
    float p = (k < nk16) ? srow[k] * inv : 0.0f;
    arow[k] = p;                          // fp32 att output (zeros in masked tail)
    if (k < kvpad) prow[k] = f2bf(p);     // bf16 copy for P*V
  }
  __syncthreads();

  // -------- phase 3: Y = P V  (waves: 4 N-subtiles x 2 K-halves), Vt = [d][t]
  const int nt = w & 3, g = w >> 2;
  const int nchunks = kvpad >> 5;
  v8f acc = vzero();
  bf16* vw = vs + w * (16 * VSTRIDE);
  for (int c = g; c < nchunks; c += 2) {
    { // stage Vt tile [16 d][32 k] (32B per lane, contiguous both sides)
      int r = lane >> 1, off = (lane & 1) * 16;
      async_b128x2(&Vp[(size_t)(nt * 16 + r) * T_ + c * 32 + off], &vw[r * VSTRIDE + off]);
    }
    wait_async0();
    Frag A, Bf;
#pragma unroll
    for (int i = 0; i < 8; ++i) {
      int ka = (i < 4 ? 2 * i : 2 * i + 8) + halfA * 8 + c * 32;
      A.u[i] = *(const unsigned*)&pb[mn * PBS + ka];
      int kb = (i < 4 ? 2 * i : 2 * i + 8) + halfA * 8;
      Bf.u[i] = *(const unsigned*)&vw[mn * VSTRIDE + kb];   // B[k][n] = Vt[n][k]
    }
    acc = __builtin_amdgcn_wmma_f32_16x16x32_bf16(false, A.v, false, Bf.v, (short)0, acc, false, false);
  }
  __syncthreads();
  if (w >= 4) {
#pragma unroll
    for (int r = 0; r < 8; ++r)
      part[(nt * 16 + r + 8 * halfA) * 16 + mn] = acc[r];
  }
  __syncthreads();
  if (w < 4) {
#pragma unroll
    for (int r = 0; r < 8; ++r) {
      int m = r + 8 * halfA;
      float v = acc[r] + part[(nt * 16 + m) * 16 + mn];
      Yint[((size_t)b * T_ + q0 + m) * C_ + h * D_ + nt * 16 + mn] = f2bf(v);
    }
  }
}

// ---------------------------------------------------------------- launch
extern "C" void kernel_launch(void* const* d_in, const int* in_sizes, int n_in,
                              void* d_out, int out_size, void* d_ws, size_t ws_size,
                              hipStream_t stream) {
  (void)in_sizes; (void)n_in; (void)out_size; (void)ws_size;
  const float* x  = (const float*)d_in[0];
  const float* Wq = (const float*)d_in[1];
  const float* bq = (const float*)d_in[2];
  const float* Wk = (const float*)d_in[3];
  const float* bk = (const float*)d_in[4];
  const float* Wv = (const float*)d_in[5];
  const float* bv = (const float*)d_in[6];
  const float* Wp = (const float*)d_in[7];
  const float* bp = (const float*)d_in[8];

  float* y_out   = (float*)d_out;
  float* att_out = y_out + (size_t)B_ * T_ * C_;

  char* ws = (char*)d_ws;
  const size_t MB = 1u << 20;
  bf16* xb  = (bf16*)(ws);              // 16 MB
  bf16* Wqb = (bf16*)(ws + 16 * MB);    //  2 MB each, TRANSPOSED [n][k]
  bf16* Wkb = (bf16*)(ws + 18 * MB);
  bf16* Wvb = (bf16*)(ws + 20 * MB);
  bf16* Wpb = (bf16*)(ws + 22 * MB);
  bf16* Qb  = (bf16*)(ws + 24 * MB);    // [B,H,T,D] 16 MB
  bf16* Kb  = (bf16*)(ws + 40 * MB);    // [B,H,T,D]
  bf16* Vb  = (bf16*)(ws + 56 * MB);    // [B,H,D,T]
  bf16* Yi  = (bf16*)(ws + 72 * MB);    // [B,T,C] bf16 16 MB

  const int NX = B_ * T_ * C_;
  f32_to_bf16_kernel<<<(NX + 255) / 256, 256, 0, stream>>>(x, xb, NX);
  dim3 tgrid(C_ / 32, C_ / 32);
  f32_to_bf16_T_kernel<<<tgrid, 256, 0, stream>>>(Wq, Wqb);
  f32_to_bf16_T_kernel<<<tgrid, 256, 0, stream>>>(Wk, Wkb);
  f32_to_bf16_T_kernel<<<tgrid, 256, 0, stream>>>(Wv, Wvb);
  f32_to_bf16_T_kernel<<<tgrid, 256, 0, stream>>>(Wp, Wpb);

  qkv_gemm_kernel<<<dim3(C_ / BN, (B_ * T_) / BM, 3), 256, 0, stream>>>(
      xb, Wqb, Wkb, Wvb, bq, bk, bv, Qb, Kb, Vb);

  attn_kernel<<<dim3(T_ / 16, H_, B_), 256, ATTN_SMEM, stream>>>(
      Qb, Kb, Vb, att_out, Yi);

  proj_gemm_kernel<<<dim3(C_ / BN, (B_ * T_) / BM), 256, 0, stream>>>(
      Yi, Wpb, bp, y_out);
}
